// GPR_59399397703844
// MI455X (gfx1250) — matrix operand order repaired and checked by reading, not compiled, not run on hardware
//
#include <hip/hip_runtime.h>
#include <math.h>

// ---------------- problem constants ----------------
constexpr int CB = 2, CL = 2048, CD = 1024, CH = 16, CDK = 64, CT = 4, CF = 4096;
constexpr int CNTOK = CB * CL; // 4096

// ---------------- WMMA types / helpers ----------------
typedef __attribute__((ext_vector_type(16))) __bf16 v16bf;
typedef __attribute__((ext_vector_type(8)))  __bf16 v8bf;
typedef __attribute__((ext_vector_type(8)))  float  v8f;

static __device__ __forceinline__ v8f wmma_bf16(v16bf a, v16bf b, v8f c) {
  // D = A(16x32 bf16) x B(32x16 bf16) + C(16x16 f32)
  return __builtin_amdgcn_wmma_f32_16x16x32_bf16(false, a, false, b, (short)0, c, false, false);
}
static __device__ __forceinline__ v8bf ld8(const __bf16* p) {
  return *reinterpret_cast<const v8bf*>(p);
}
static __device__ __forceinline__ v16bf cmb(v8bf lo, v8bf hi) {
  v16bf r;
#pragma unroll
  for (int i = 0; i < 8; ++i) { r[i] = lo[i]; r[i + 8] = hi[i]; }
  return r;
}
static __device__ __forceinline__ v8f zero8() {
  v8f z;
#pragma unroll
  for (int i = 0; i < 8; ++i) z[i] = 0.f;
  return z;
}

// ---------------- CDNA5 async global->LDS copy (ASYNCcnt-tracked DMA) ----------------
// Wave-relative LDS byte address = low 32 bits of the generic pointer (ISA 10.2).
static __device__ __forceinline__ uint32_t lds_addr(const void* p) {
  return (uint32_t)(uintptr_t)p;
}
static __device__ __forceinline__ void async_ld_b128(uint32_t lds_byte_addr, const void* gsrc) {
  asm volatile("global_load_async_to_lds_b128 %0, %1, off"
               :: "v"(lds_byte_addr), "v"(gsrc)
               : "memory");
}
static __device__ __forceinline__ void wait_async0() {
  asm volatile("s_wait_asynccnt 0x0" ::: "memory");
}

// ---------------- fp32 -> bf16 conversion ----------------
__global__ void cvt_bf16_kernel(const float* __restrict__ s, __bf16* __restrict__ d, int n) {
  int i = blockIdx.x * 256 + threadIdx.x;
  if (i < n) d[i] = (__bf16)s[i];
}

// ---------------- per-type token index (gather lists) ----------------
__global__ void zero_cnt_kernel(int* cnt) {
  if (threadIdx.x < CT) cnt[threadIdx.x] = 0;
}
__global__ void build_idx_kernel(const int* __restrict__ tt, int* cnt, int* idx) {
  int i = blockIdx.x * 256 + threadIdx.x;
  if (i < CNTOK) {
    int t = tt[i];
    int s = atomicAdd(&cnt[t], 1);
    idx[t * CNTOK + s] = i;
  }
}

// ---------------- generic WMMA GEMM (double-buffered, async A staging) ----------------
// C[M,Nn] = A[M,Kd](bf16) x Bw[Kd,Nn](bf16), tiles 64x64, 4 waves/block.
// EP: 0 = store bf16; 1 = +bias, exact GELU, store bf16; 2 = +bias +resid, store f32
// GATHER: row index map gidx[] for A rows and C rows, M clamped by *cntp.
template <int EP, bool GATHER>
__global__ __launch_bounds__(128)
void wmma_gemm_kernel(const __bf16* __restrict__ A, const __bf16* __restrict__ Bw,
                      const float* __restrict__ bias, const float* __restrict__ resid,
                      float* __restrict__ outF, __bf16* __restrict__ outB,
                      const int* __restrict__ gidx, const int* __restrict__ cntp,
                      int Mfull, int Nn, int Kd, int lda, int ldb) {
  int M = Mfull;
  if (GATHER) { int c = cntp[0]; M = c < Mfull ? c : Mfull; }
  int m0 = blockIdx.y * 64;
  int n0 = blockIdx.x * 64;
  if (m0 >= M) return;

  __shared__ __align__(16) __bf16 la[2][64][40];   // A tile, row-major [m][k]
  __shared__ __align__(16) __bf16 lbT[2][64][40];  // B tile, transposed [n][k]

  int tid = threadIdx.x;
  int lane = tid & 31, w = tid >> 5;
  int hl = lane >> 4, l16 = lane & 15;
  int koff = hl * 8, khalf = hl * 16;

  // K-invariant staging bases (gather index hoisted out of the K loop)
  int ar_ = tid >> 1, hf_ = tid & 1;               // A: 64 rows x 2 halves
  int slot = m0 + ar_;
  if (GATHER) { if (slot >= M) slot = M - 1; }
  int grow = GATHER ? gidx[slot] : slot;
  const __bf16* asrc = A + (size_t)grow * lda + hf_ * 16;
  uint32_t adst0 = lds_addr(&la[0][ar_][hf_ * 16]);
  uint32_t adst1 = lds_addr(&la[1][ar_][hf_ * 16]);

  int kr_ = tid >> 2, qn_ = tid & 3;               // B: 32 k-rows x 4 n-quarters
  const __bf16* bsrc = Bw + (size_t)kr_ * ldb + n0 + qn_ * 16;

  v8f acc[4];
#pragma unroll
  for (int i = 0; i < 4; ++i) acc[i] = zero8();

  auto stageA = [&](int k0, int b) {
    uint32_t d = b ? adst1 : adst0;
    const __bf16* s = asrc + k0;
    async_ld_b128(d, s);
    async_ld_b128(d + 16, s + 8);
  };
  auto stageB = [&](int k0, int b) {
    const __bf16* s = bsrc + (size_t)k0 * ldb;
    v8bf c0 = ld8(s), c1 = ld8(s + 8);
#pragma unroll
    for (int j = 0; j < 8; ++j) {
      lbT[b][qn_ * 16 + j][kr_]     = c0[j];
      lbT[b][qn_ * 16 + 8 + j][kr_] = c1[j];
    }
    __builtin_prefetch(s + (size_t)32 * ldb, 0, 3);
  };

  stageA(0, 0);
  stageB(0, 0);
  wait_async0();
  __syncthreads();

  int cur = 0;
  for (int k0 = 0; k0 < Kd; k0 += 32) {
    int nxt = cur ^ 1;
    if (k0 + 32 < Kd) {            // prefetch next tile while computing this one
      stageA(k0 + 32, nxt);
      stageB(k0 + 32, nxt);
    }
    int ar = (w << 4) | l16;
    v16bf af = cmb(ld8(&la[cur][ar][khalf]), ld8(&la[cur][ar][khalf + 8]));
#pragma unroll
    for (int nt = 0; nt < 4; ++nt) {
      int bc = nt * 16 + l16;
      v16bf bf = cmb(ld8(&lbT[cur][bc][khalf]), ld8(&lbT[cur][bc][khalf + 8]));
      acc[nt] = wmma_bf16(af, bf, acc[nt]);
    }
    wait_async0();                 // async copy finished (overlapped with WMMAs)
    __syncthreads();
    cur = nxt;
  }

  // epilogue (C layout: lane l16 = column, VGPR v = row v + 8*hl)
#pragma unroll
  for (int nt = 0; nt < 4; ++nt) {
#pragma unroll
    for (int v = 0; v < 8; ++v) {
      int oslot = m0 + (w << 4) + v + 8 * hl;
      if (oslot >= M) continue;
      int r = GATHER ? gidx[oslot] : oslot;
      int c = n0 + nt * 16 + l16;
      float xv = acc[nt][v];
      if (bias) xv += bias[c];
      if (EP == 1) xv = 0.5f * xv * (1.0f + erff(xv * 0.70710678118654752f));
      size_t o = (size_t)r * Nn + c;
      if (EP == 2) { if (resid) xv += resid[o]; outF[o] = xv; }
      else outB[o] = (__bf16)xv;
    }
  }
}

// ---------------- HSTU flash attention (hybrid mask + padding + U gate) ----------------
// 4 waves / block: 64 query rows share cooperatively staged, double-buffered K/V tiles.
__global__ __launch_bounds__(128)
void hstu_attn_kernel(const __bf16* __restrict__ Q, const __bf16* __restrict__ K,
                      const __bf16* __restrict__ V, const __bf16* __restrict__ U,
                      const int* __restrict__ tt, const int* __restrict__ slp,
                      __bf16* __restrict__ outg) {
  int m0b = blockIdx.x << 6;  // 64 query rows per block
  int h   = blockIdx.y;
  int bz  = blockIdx.z;

  __shared__ __align__(16) __bf16 lk[2][32][72];   // K tile [key][dk]
  __shared__ __align__(16) __bf16 lvT[2][64][40];  // V tile transposed [dk][key]
  __shared__ __align__(16) __bf16 lp[4][16][40];   // per-wave P tile [q][key]

  int tid = threadIdx.x;
  int w = tid >> 5, lane = tid & 31, hl = lane >> 4, l16 = lane & 15;
  int koff = hl * 8, khalf = hl * 16;
  int m0 = m0b + (w << 4);
  int slen = slp[bz];

  // per-wave Q fragments (16 rows x 64 dk = two K-steps)
  size_t qbase = ((size_t)(bz * CL + m0 + l16)) * CD + (size_t)h * CDK;
  v16bf aq0 = cmb(ld8(Q + qbase + khalf),      ld8(Q + qbase + khalf + 8));
  v16bf aq1 = cmb(ld8(Q + qbase + 32 + khalf), ld8(Q + qbase + 32 + khalf + 8));

  // cooperative staging partition: 32 key-rows x 4 dk-quarters
  int kr_ = tid >> 2, qn_ = tid & 3;
  const __bf16* kbase = K + (size_t)(bz * CL) * CD + (size_t)h * CDK;
  const __bf16* vbase = V + (size_t)(bz * CL) * CD + (size_t)h * CDK;
  uint32_t kdst0 = lds_addr(&lk[0][kr_][qn_ * 16]);
  uint32_t kdst1 = lds_addr(&lk[1][kr_][qn_ * 16]);

  auto stageKV = [&](int key0, int b) {
    const __bf16* ks = kbase + (size_t)(key0 + kr_) * CD + qn_ * 16;
    uint32_t kd = b ? kdst1 : kdst0;
    async_ld_b128(kd, ks);                  // K tile: ASYNCcnt DMA, row-major
    async_ld_b128(kd + 16, ks + 8);
    const __bf16* vs = vbase + (size_t)(key0 + kr_) * CD + qn_ * 16;
    v8bf c0 = ld8(vs), c1 = ld8(vs + 8);    // V tile: manual transpose scatter
#pragma unroll
    for (int j = 0; j < 8; ++j) {
      lvT[b][qn_ * 16 + j][kr_]     = c0[j];
      lvT[b][qn_ * 16 + 8 + j][kr_] = c1[j];
    }
  };

  v8f o0 = zero8(), o1 = zero8(), o2 = zero8(), o3 = zero8();
  float rm[8], rl[8];
#pragma unroll
  for (int v = 0; v < 8; ++v) { rm[v] = -1e30f; rl[v] = 0.f; }

  stageKV(0, 0);
  wait_async0();
  __syncthreads();

  int cur = 0;
  for (int key0 = 0; key0 < CL; key0 += 32) {
    int nxt = cur ^ 1;
    if (key0 + 32 < CL) stageKV(key0 + 32, nxt);   // overlap with compute

    // S = Q x K^T  (16 q rows x 32 keys), 2 N-tiles x 2 K-steps
    v8f s0 = zero8(), s1 = zero8();
    {
      v16bf b00 = cmb(ld8(&lk[cur][l16][khalf]),           ld8(&lk[cur][l16][khalf + 8]));
      v16bf b01 = cmb(ld8(&lk[cur][16 + l16][khalf]),      ld8(&lk[cur][16 + l16][khalf + 8]));
      s0 = wmma_bf16(aq0, b00, s0);
      s1 = wmma_bf16(aq0, b01, s1);
      v16bf b10 = cmb(ld8(&lk[cur][l16][32 + khalf]),      ld8(&lk[cur][l16][32 + khalf + 8]));
      v16bf b11 = cmb(ld8(&lk[cur][16 + l16][32 + khalf]), ld8(&lk[cur][16 + l16][32 + khalf + 8]));
      s0 = wmma_bf16(aq1, b10, s0);
      s1 = wmma_bf16(aq1, b11, s1);
    }

    // mask + online softmax.  C layout: lane = key column, VGPR v = q row v+8*hl.
    int j0 = key0 + l16, j1 = key0 + 16 + l16;
    bool pr0 = tt[bz * CL + j0] < 3, pr1 = tt[bz * CL + j1] < 3;
    bool pad0 = j0 >= slen, pad1 = j1 >= slen;

    float al[8];
#pragma unroll
    for (int v = 0; v < 8; ++v) {
      int qpos = m0 + v + 8 * hl;
      float x0 = s0[v] * 0.125f;  // 1/sqrt(64)
      float x1 = s1[v] * 0.125f;
      if (pad0 || !(pr0 || qpos >= j0)) x0 = -1e30f;
      if (pad1 || !(pr1 || qpos >= j1)) x1 = -1e30f;
      float tm = fmaxf(x0, x1);
      tm = fmaxf(tm, __shfl_xor(tm, 1, 16));
      tm = fmaxf(tm, __shfl_xor(tm, 2, 16));
      tm = fmaxf(tm, __shfl_xor(tm, 4, 16));
      tm = fmaxf(tm, __shfl_xor(tm, 8, 16));
      float mn = fmaxf(rm[v], tm);
      float a  = __expf(rm[v] - mn);
      rm[v] = mn;
      float p0 = __expf(x0 - mn), p1 = __expf(x1 - mn);
      float rs = p0 + p1;
      rs += __shfl_xor(rs, 1, 16);
      rs += __shfl_xor(rs, 2, 16);
      rs += __shfl_xor(rs, 4, 16);
      rs += __shfl_xor(rs, 8, 16);
      rl[v] = rl[v] * a + rs;
      al[v] = a;
      lp[w][v + 8 * hl][l16]      = (__bf16)p0;  // re-layout P through LDS (per-wave,
      lp[w][v + 8 * hl][16 + l16] = (__bf16)p1;  // same-wave DS ops are in-order)
    }
#pragma unroll
    for (int v = 0; v < 8; ++v) { o0[v] *= al[v]; o1[v] *= al[v]; o2[v] *= al[v]; o3[v] *= al[v]; }

    // O += P(16x32) x V(32x64)
    v16bf ap = cmb(ld8(&lp[w][l16][khalf]), ld8(&lp[w][l16][khalf + 8]));
    o0 = wmma_bf16(ap, cmb(ld8(&lvT[cur][l16][khalf]),      ld8(&lvT[cur][l16][khalf + 8])),      o0);
    o1 = wmma_bf16(ap, cmb(ld8(&lvT[cur][16 + l16][khalf]), ld8(&lvT[cur][16 + l16][khalf + 8])), o1);
    o2 = wmma_bf16(ap, cmb(ld8(&lvT[cur][32 + l16][khalf]), ld8(&lvT[cur][32 + l16][khalf + 8])), o2);
    o3 = wmma_bf16(ap, cmb(ld8(&lvT[cur][48 + l16][khalf]), ld8(&lvT[cur][48 + l16][khalf + 8])), o3);

    wait_async0();                 // next K tile landed (overlapped)
    __syncthreads();               // all waves done with cur before restage
    cur = nxt;
  }

  // normalize + U gate + store bf16
#pragma unroll
  for (int v = 0; v < 8; ++v) {
    int tok = bz * CL + m0 + v + 8 * hl;
    float inv = 1.0f / rl[v];
    size_t base = (size_t)tok * CD + (size_t)h * CDK;
    float u0 = (float)U[base + l16];
    float u1 = (float)U[base + 16 + l16];
    float u2 = (float)U[base + 32 + l16];
    float u3 = (float)U[base + 48 + l16];
    outg[base + l16]      = (__bf16)(o0[v] * inv * u0);
    outg[base + 16 + l16] = (__bf16)(o1[v] * inv * u1);
    outg[base + 32 + l16] = (__bf16)(o2[v] * inv * u2);
    outg[base + 48 + l16] = (__bf16)(o3[v] * inv * u3);
  }
}

// ---------------- per-token LayerNorm with per-type gamma/beta ----------------
__global__ __launch_bounds__(256)
void token_ln_kernel(const float* __restrict__ xin, const float* __restrict__ g,
                     const float* __restrict__ bta, const int* __restrict__ tt,
                     float* __restrict__ outF, __bf16* __restrict__ outB) {
  int tok = blockIdx.x;
  int tid = threadIdx.x;
  const float* row = xin + (size_t)tok * CD;
  __shared__ float red[256];

  float s = 0.f;
  for (int i = tid; i < CD; i += 256) s += row[i];
  red[tid] = s; __syncthreads();
  for (int o = 128; o > 0; o >>= 1) { if (tid < o) red[tid] += red[tid + o]; __syncthreads(); }
  float mu = red[0] * (1.0f / CD);
  __syncthreads();

  float s2 = 0.f;
  for (int i = tid; i < CD; i += 256) { float d = row[i] - mu; s2 += d * d; }
  red[tid] = s2; __syncthreads();
  for (int o = 128; o > 0; o >>= 1) { if (tid < o) red[tid] += red[tid + o]; __syncthreads(); }
  float rstd = rsqrtf(red[0] * (1.0f / CD) + 1e-5f);

  int t = tt[tok];
  const float* gg = g + (size_t)t * CD;
  const float* bb = bta + (size_t)t * CD;
  for (int i = tid; i < CD; i += 256) {
    float y = (row[i] - mu) * rstd * gg[i] + bb[i];
    outF[(size_t)tok * CD + i] = y;
    if (outB) outB[(size_t)tok * CD + i] = (__bf16)y;
  }
}

// ---------------- host orchestration ----------------
extern "C" void kernel_launch(void* const* d_in, const int* in_sizes, int n_in,
                              void* d_out, int out_size, void* d_ws, size_t ws_size,
                              hipStream_t stream) {
  const float* x    = (const float*)d_in[0];
  const int*   tt   = (const int*)d_in[1];
  const int*   sl   = (const int*)d_in[2];
  const float* Wq   = (const float*)d_in[3];
  const float* Wk   = (const float*)d_in[4];
  const float* Wv   = (const float*)d_in[5];
  const float* Wu   = (const float*)d_in[6];
  const float* Wo   = (const float*)d_in[7];
  const float* ln1g = (const float*)d_in[8];
  const float* ln1b = (const float*)d_in[9];
  const float* W1   = (const float*)d_in[10];
  const float* b1   = (const float*)d_in[11];
  const float* W2   = (const float*)d_in[12];
  const float* b2   = (const float*)d_in[13];
  const float* ln2g = (const float*)d_in[14];
  const float* ln2b = (const float*)d_in[15];

  char* ws = (char*)d_ws;
  size_t off = 0;
  auto alloc = [&](size_t bytes) -> char* {
    char* p = ws + off;
    off = (off + bytes + 255) & ~(size_t)255;
    return p;
  };

  __bf16* XB   = (__bf16*)alloc((size_t)CNTOK * CD * 2);
  __bf16* WQB  = (__bf16*)alloc((size_t)CD * CD * 2);
  __bf16* WKB  = (__bf16*)alloc((size_t)CD * CD * 2);
  __bf16* WVB  = (__bf16*)alloc((size_t)CD * CD * 2);
  __bf16* WUB  = (__bf16*)alloc((size_t)CD * CD * 2);
  __bf16* WOB  = (__bf16*)alloc((size_t)CD * CD * 2);
  __bf16* W1B  = (__bf16*)alloc((size_t)CT * CD * CF * 2);
  __bf16* W2B  = (__bf16*)alloc((size_t)CT * CF * CD * 2);
  __bf16* QB   = (__bf16*)alloc((size_t)CNTOK * CD * 2);
  __bf16* KB   = (__bf16*)alloc((size_t)CNTOK * CD * 2);
  __bf16* VB   = (__bf16*)alloc((size_t)CNTOK * CD * 2);
  __bf16* UB   = (__bf16*)alloc((size_t)CNTOK * CD * 2);
  __bf16* ATTG = (__bf16*)alloc((size_t)CNTOK * CD * 2);
  float*  APROJ= (float*) alloc((size_t)CNTOK * CD * 4);
  float*  X1   = (float*) alloc((size_t)CNTOK * CD * 4);
  __bf16* X1B  = (__bf16*)alloc((size_t)CNTOK * CD * 2);
  __bf16* HMID = (__bf16*)alloc((size_t)CNTOK * CF * 2);
  float*  X2P  = (float*) alloc((size_t)CNTOK * CD * 4);
  int*    CNTP = (int*)   alloc(CT * sizeof(int));
  int*    IDXP = (int*)   alloc((size_t)CT * CNTOK * sizeof(int));
  (void)ws_size; (void)in_sizes; (void)n_in; (void)out_size;

  auto cvt = [&](const float* s, __bf16* dp, int n) {
    cvt_bf16_kernel<<<(n + 255) / 256, 256, 0, stream>>>(s, dp, n);
  };
  cvt(x,  XB,  CNTOK * CD);
  cvt(Wq, WQB, CD * CD);
  cvt(Wk, WKB, CD * CD);
  cvt(Wv, WVB, CD * CD);
  cvt(Wu, WUB, CD * CD);
  cvt(Wo, WOB, CD * CD);
  cvt(W1, W1B, CT * CD * CF);
  cvt(W2, W2B, CT * CF * CD);

  zero_cnt_kernel<<<1, 32, 0, stream>>>(CNTP);
  build_idx_kernel<<<CNTOK / 256, 256, 0, stream>>>(tt, CNTP, IDXP);

  // Q/K/V/U projections: [4096,1024] x [1024,1024] -> bf16
  dim3 gProj(CD / 64, CNTOK / 64);
  wmma_gemm_kernel<0, false><<<gProj, 128, 0, stream>>>(
      XB, WQB, nullptr, nullptr, nullptr, QB, nullptr, nullptr, CNTOK, CD, CD, CD, CD);
  wmma_gemm_kernel<0, false><<<gProj, 128, 0, stream>>>(
      XB, WKB, nullptr, nullptr, nullptr, KB, nullptr, nullptr, CNTOK, CD, CD, CD, CD);
  wmma_gemm_kernel<0, false><<<gProj, 128, 0, stream>>>(
      XB, WVB, nullptr, nullptr, nullptr, VB, nullptr, nullptr, CNTOK, CD, CD, CD, CD);
  wmma_gemm_kernel<0, false><<<gProj, 128, 0, stream>>>(
      XB, WUB, nullptr, nullptr, nullptr, UB, nullptr, nullptr, CNTOK, CD, CD, CD, CD);

  // flash attention with hybrid mask + U gate (4 q-tiles share K/V per block)
  hstu_attn_kernel<<<dim3(CL / 64, CH, CB), 128, 0, stream>>>(QB, KB, VB, UB, tt, sl, ATTG);

  // output projection + residual: APROJ = ATTG@Wo + x
  wmma_gemm_kernel<2, false><<<gProj, 128, 0, stream>>>(
      ATTG, WOB, nullptr, x, APROJ, nullptr, nullptr, nullptr, CNTOK, CD, CD, CD, CD);

  // LN1 -> X1 (f32) + X1B (bf16)
  token_ln_kernel<<<CNTOK, 256, 0, stream>>>(APROJ, ln1g, ln1b, tt, X1, X1B);

  // type-gathered FFN: only each token's own expert weights
  for (int t = 0; t < CT; ++t) {
    wmma_gemm_kernel<1, true><<<dim3(CF / 64, CNTOK / 64), 128, 0, stream>>>(
        X1B, W1B + (size_t)t * CD * CF, b1 + (size_t)t * CF, nullptr,
        nullptr, HMID, IDXP + (size_t)t * CNTOK, CNTP + t,
        CNTOK, CF, CD, CD, CF);
  }
  for (int t = 0; t < CT; ++t) {
    wmma_gemm_kernel<2, true><<<dim3(CD / 64, CNTOK / 64), 128, 0, stream>>>(
        HMID, W2B + (size_t)t * CF * CD, b2 + (size_t)t * CD, X1,
        X2P, nullptr, IDXP + (size_t)t * CNTOK, CNTP + t,
        CNTOK, CD, CF, CF, CD);
  }

  // LN2 -> final output (f32)
  token_ln_kernel<<<CNTOK, 256, 0, stream>>>(X2P, ln2g, ln2b, tt, (float*)d_out, nullptr);
}
